// S4_86036784873609
// MI455X (gfx1250) — matrix operand order repaired and checked by reading, not compiled, not run on hardware
//
#include <hip/hip_runtime.h>
#include <hip/hip_bf16.h>
#include <math.h>

// Problem constants (B,H,L) = (4,1024,4096), FFT length 2L = 8192.
#define BZ   4
#define HH   1024
#define LL   4096
#define NFFT 8192
#define NFH  4097      // Hermitian half bins stored
#define KFP  4104      // padded pitch (complex) per h
#define NTOT (BZ*LL)   // 16384 = GEMM N

typedef __attribute__((ext_vector_type(16))) __bf16 v16bf;
typedef __attribute__((ext_vector_type(8)))  float  v8f;
typedef __attribute__((ext_vector_type(4)))  int    v4i;

#if defined(__has_builtin)
#if __has_builtin(__builtin_amdgcn_global_load_async_to_lds_b128)
#define HAVE_ASYNC_LDS 1
#endif
#if __has_builtin(__builtin_amdgcn_s_wait_asynccnt)
#define HAVE_WAIT_ASYNC_BUILTIN 1
#endif
#endif

static __device__ __forceinline__ void wait_async0() {
#if defined(HAVE_WAIT_ASYNC_BUILTIN)
  __builtin_amdgcn_s_wait_asynccnt(0);
#elif defined(HAVE_ASYNC_LDS)
  asm volatile("s_wait_asynccnt 0x0" ::: "memory");
#endif
}

static __device__ __forceinline__ unsigned short f2bf(float f) {
  unsigned int u = __float_as_uint(f);
  u += 0x7FFFu + ((u >> 16) & 1u);          // round-to-nearest-even
  return (unsigned short)(u >> 16);
}

// In-place radix-2 DIT FFT, N=8192, data + twiddle table in LDS.
// Table holds W_N^j = exp(-2*pi*i*j/N), j in [0, N/2). inv=true conjugates.
static __device__ void fft8192_lds(float* re, float* im,
                                   const float* twr, const float* twi,
                                   int tid, int nthr, bool inv) {
  // bit-reverse permutation (13 bits)
  for (int t = tid; t < NFFT; t += nthr) {
    int j = (int)(__brev((unsigned)t) >> 19);
    if (j > t) {
      float a = re[t]; re[t] = re[j]; re[j] = a;
      float b = im[t]; im[t] = im[j]; im[j] = b;
    }
  }
  __syncthreads();
  for (int l2h = 0; l2h < 13; ++l2h) {
    int half = 1 << l2h;
    for (int idx = tid; idx < NFFT / 2; idx += nthr) {
      int pos = idx & (half - 1);
      int i   = ((idx >> l2h) << (l2h + 1)) + pos;
      int j   = i + half;
      int tw  = pos << (12 - l2h);
      float c = twr[tw];
      float s = inv ? -twi[tw] : twi[tw];
      float xr = re[j], xi = im[j];
      float tr = xr * c - xi * s;
      float ti = xr * s + xi * c;
      float ur = re[i], ui = im[i];
      re[j] = ur - tr; im[j] = ui - ti;
      re[i] = ur + tr; im[i] = ui + ti;
    }
    __syncthreads();
  }
}

static __device__ __forceinline__ void build_twiddles(float* twr, float* twi, int tid, int nthr) {
  const float w = -6.2831853071795864769f / (float)NFFT;
  for (int j = tid; j < NFFT / 2; j += nthr) {
    float s, c;
    __sincosf(w * (float)j, &s, &c);
    twr[j] = c; twi[j] = s;
  }
}

// ---------------- Kernel 1: k_f = rfft(pad(k0) + pad(rev(k1))), per h ----------------
__global__ __launch_bounds__(512) void s4_kf_kernel(const float* __restrict__ k,
                                                    float2* __restrict__ kf) {
  __shared__ float re[NFFT];
  __shared__ float im[NFFT];
  __shared__ float twr[NFFT / 2];
  __shared__ float twi[NFFT / 2];
  int h = blockIdx.x, tid = threadIdx.x, nthr = blockDim.x;
  build_twiddles(twr, twi, tid, nthr);
  const float* k0p = k + (size_t)h * LL;
  const float* k1p = k + (size_t)HH * LL + (size_t)h * LL;
  for (int t = tid; t < LL; t += nthr) {
    re[t]      = k0p[t];
    re[LL + t] = k1p[LL - 1 - t];
    im[t] = 0.f; im[LL + t] = 0.f;
  }
  __syncthreads();
  fft8192_lds(re, im, twr, twi, tid, nthr, false);
  for (int f = tid; f < NFH; f += nthr)
    kf[(size_t)h * KFP + f] = make_float2(re[f], im[f]);
}

// ---------------- Kernel 2: per (b,h): FFT conv + D*u + gelu -> bf16 act ----------------
__global__ __launch_bounds__(512) void s4_conv_gelu_kernel(const float* __restrict__ u,
                                                           const float* __restrict__ D,
                                                           const float2* __restrict__ kf,
                                                           unsigned short* __restrict__ act) {
  __shared__ float re[NFFT];
  __shared__ float im[NFFT];
  __shared__ float twr[NFFT / 2];
  __shared__ float twi[NFFT / 2];
  int h = blockIdx.x, b = blockIdx.y, tid = threadIdx.x, nthr = blockDim.x;
  build_twiddles(twr, twi, tid, nthr);
  const float* up = u + ((size_t)b * HH + h) * LL;
  for (int t = tid; t < LL; t += nthr) {
    re[t] = up[t]; re[LL + t] = 0.f;
    im[t] = 0.f;   im[LL + t] = 0.f;
  }
  __syncthreads();
  fft8192_lds(re, im, twr, twi, tid, nthr, false);     // u_f
  const float2* kfh = kf + (size_t)h * KFP;
  for (int f = tid; f < NFFT; f += nthr) {             // y_f = u_f * k_f (Hermitian extend)
    float2 kv;
    if (f <= LL) kv = kfh[f];
    else { float2 c = kfh[NFFT - f]; kv = make_float2(c.x, -c.y); }
    float ur = re[f], ui = im[f];
    re[f] = ur * kv.x - ui * kv.y;
    im[f] = ur * kv.y + ui * kv.x;
  }
  __syncthreads();
  fft8192_lds(re, im, twr, twi, tid, nthr, true);      // unscaled irfft
  float dh = D[h];
  const float inv = 1.0f / (float)NFFT;
  unsigned short* ap = act + (size_t)h * NTOT + (size_t)b * LL;
  for (int t = tid; t < LL; t += nthr) {
    float y = re[t] * inv + dh * up[t];
    float g = 0.5f * y * (1.0f + erff(y * 0.70710678118654752f));  // exact gelu
    ap[t] = f2bf(g);
  }
}

// ---------------- Kernel 3a: W f32 -> bf16 ----------------
__global__ __launch_bounds__(256) void s4_wcvt_kernel(const float* __restrict__ W,
                                                      unsigned short* __restrict__ Wbf, int n) {
  int i = blockIdx.x * blockDim.x + threadIdx.x;
  if (i < n) Wbf[i] = f2bf(W[i]);
}

// ---------------- Kernel 3b: out[b,o,t] = sum_h W[o,h]*act[h,(b,t)] + bias[o] ----------------
// bf16 WMMA GEMM: M=1024 (o), N=16384 (b*t), K=1024 (h). 128x128x32 tiles, 8 waves/block.
__global__ __launch_bounds__(256) void s4_gemm_kernel(const unsigned short* __restrict__ A,   // Wbf [o][h]
                                                      const unsigned short* __restrict__ Bm,  // act [h][n]
                                                      const float* __restrict__ bias,
                                                      float* __restrict__ out) {
  __shared__ unsigned short As[128 * 32];   // [row(o)][k]
  __shared__ unsigned short Bs[128 * 32];   // transposed: [col(n)][k]
  int tid = threadIdx.x, lane = tid & 31, wave = tid >> 5;
  int m0 = blockIdx.y * 128, n0 = blockIdx.x * 128;
  int wm = wave & 3;    // 32-row strip of M
  int wn = wave >> 2;   // 64-col strip of N
  v8f zero = {};
  v8f acc[2][4];
  for (int i = 0; i < 2; i++)
    for (int j = 0; j < 4; j++) acc[i][j] = zero;

  int arow  = lane & 15;
  int akoff = (lane >> 4) << 3;   // A: K group offset 0/8 (16-bit A layout)
  int bcol  = lane & 15;
  int bkoff = (lane >> 4) << 3;   // B: K group offset 0/8

  for (int k0 = 0; k0 < 1024; k0 += 32) {
    // stage A tile (128x32): layout matches global, use async global->LDS when available
#if defined(HAVE_ASYNC_LDS)
    for (int v = tid; v < 512; v += 256) {
      int row = v >> 2, c8 = (v & 3) << 3;
      const unsigned short* g = A + (size_t)(m0 + row) * 1024 + k0 + c8;
      __builtin_amdgcn_global_load_async_to_lds_b128(
          (v4i*)g, (v4i*)&As[row * 32 + c8], 0, 0);
    }
#else
    for (int v = tid; v < 512; v += 256) {
      int row = v >> 2, c8 = (v & 3) << 3;
      uint4 d = *(const uint4*)(A + (size_t)(m0 + row) * 1024 + k0 + c8);
      *(uint4*)(&As[row * 32 + c8]) = d;
    }
#endif
    // stage B tile (32x128) with in-LDS transpose -> Bs[n][k] (needs the register bounce)
    for (int v = tid; v < 512; v += 256) {
      int kr = v >> 4, c8 = (v & 15) << 3;
      uint4 d = *(const uint4*)(Bm + (size_t)(k0 + kr) * NTOT + n0 + c8);
      const unsigned short* e = (const unsigned short*)&d;
#pragma unroll
      for (int j = 0; j < 8; j++) Bs[(c8 + j) * 32 + kr] = e[j];
    }
    if (k0 + 32 < 1024) {  // prefetch next K tile (global_prefetch_b8)
      __builtin_prefetch(A + (size_t)(m0 + (tid >> 1)) * 1024 + k0 + 32, 0, 0);
      __builtin_prefetch(Bm + (size_t)(k0 + 32 + (tid >> 4)) * NTOT + n0 + ((tid & 15) << 3), 0, 0);
    }
    wait_async0();
    __syncthreads();

    // build fragments per documented CDNA5 wave32 layouts
    v16bf afrag[2];
#pragma unroll
    for (int i = 0; i < 2; i++) {
      int row = wm * 32 + i * 16 + arow;
      union { unsigned short s[16]; v16bf v; } ua;
#pragma unroll
      for (int r = 0; r < 8; r++) {
        int kb = ((r & 4) << 2) + akoff + ((r & 3) << 1);  // {0,16}+{0,8}+2*(r%4)
        ua.s[2 * r]     = As[row * 32 + kb];
        ua.s[2 * r + 1] = As[row * 32 + kb + 1];
      }
      afrag[i] = ua.v;
    }
    v16bf bfrag[4];
#pragma unroll
    for (int j = 0; j < 4; j++) {
      int col = wn * 64 + j * 16 + bcol;
      union { unsigned short s[16]; v16bf v; } ub;
#pragma unroll
      for (int r = 0; r < 8; r++) {
        int kb = ((r & 4) << 2) + bkoff + ((r & 3) << 1);
        ub.s[2 * r]     = Bs[col * 32 + kb];
        ub.s[2 * r + 1] = Bs[col * 32 + kb + 1];
      }
      bfrag[j] = ub.v;
    }
#pragma unroll
    for (int i = 0; i < 2; i++)
#pragma unroll
      for (int j = 0; j < 4; j++)
        acc[i][j] = __builtin_amdgcn_wmma_f32_16x16x32_bf16(
            false, afrag[i], false, bfrag[j], (short)0, acc[i][j], false, false);
    __syncthreads();
  }

  // epilogue: D layout — VGPR r: M = r + (lane<16?0:8), N = lane%16
  int roff = (lane < 16) ? 0 : 8;
  int ccol = lane & 15;
  for (int i = 0; i < 2; i++)
    for (int j = 0; j < 4; j++) {
#pragma unroll
      for (int r = 0; r < 8; r++) {
        int o = m0 + wm * 32 + i * 16 + r + roff;
        int n = n0 + wn * 64 + j * 16 + ccol;
        int bb = n >> 12, t = n & 4095;
        out[((size_t)bb * HH + o) * LL + t] = acc[i][j][r] + bias[o];
      }
    }
}

extern "C" void kernel_launch(void* const* d_in, const int* in_sizes, int n_in,
                              void* d_out, int out_size, void* d_ws, size_t ws_size,
                              hipStream_t stream) {
  const float* u    = (const float*)d_in[0];   // (B,H,L)
  const float* k    = (const float*)d_in[1];   // (2,H,L)
  const float* D    = (const float*)d_in[2];   // (1,H)
  const float* W    = (const float*)d_in[3];   // (H,H)
  const float* bvec = (const float*)d_in[4];   // (H,)
  (void)in_sizes; (void)n_in; (void)out_size; (void)ws_size;

  char* ws = (char*)d_ws;
  size_t off = 0;
  float2* kf = (float2*)(ws + off);           off += (size_t)HH * KFP * sizeof(float2);   // 33.6 MB
  unsigned short* Wbf = (unsigned short*)(ws + off); off += (size_t)HH * HH * sizeof(unsigned short); // 2 MB
  unsigned short* act = (unsigned short*)(ws + off); off += (size_t)HH * NTOT * sizeof(unsigned short); // 32 MB

  s4_kf_kernel<<<dim3(HH), dim3(512), 0, stream>>>(k, kf);
  s4_wcvt_kernel<<<dim3((HH * HH) / 256), dim3(256), 0, stream>>>(W, Wbf, HH * HH);
  s4_conv_gelu_kernel<<<dim3(HH, BZ), dim3(512), 0, stream>>>(u, D, kf, act);
  s4_gemm_kernel<<<dim3(NTOT / 128, HH / 128), dim3(256), 0, stream>>>(Wbf, act, bvec, (float*)d_out);
}
